// TempBiasGATv2_8727373545881
// MI455X (gfx1250) — matrix operand order repaired and checked by reading, not compiled, not run on hardware
//
#include <hip/hip_runtime.h>
#include <hip/hip_fp16.h>

#define HID   128
#define HEADS 4

typedef __attribute__((ext_vector_type(16))) _Float16 v16h;
typedef __attribute__((ext_vector_type(4)))  _Float16 v4h;
typedef __attribute__((ext_vector_type(8)))  float    v8f;

// ---------------------------------------------------------------------------
// Input projection feat = relu(x@W_in+b) and z = concat(feat, emb[node_idx])
// ---------------------------------------------------------------------------
__global__ __launch_bounds__(256) void k_input(const float* __restrict__ x,
    const int* __restrict__ node_idx, const float* __restrict__ emb,
    const float* __restrict__ Win, const float* __restrict__ bin,
    float* __restrict__ z, int n) {
  int idx  = blockIdx.x * 256 + threadIdx.x;
  int node = idx >> 7, c = idx & 127;
  if (node >= n) return;
  float v;
  if (c < 64) {
    float acc = bin[c];
    const float* xr = x + (size_t)node * 16;
#pragma unroll
    for (int j = 0; j < 16; ++j) acc += xr[j] * Win[j * 64 + c];
    v = acc > 0.f ? acc : 0.f;
  } else {
    v = emb[(size_t)node_idx[node] * 64 + (c - 64)];
  }
  z[(size_t)node * HID + c] = v;
}

// ---------------------------------------------------------------------------
// Out[nrows,128] = Z[nrows,128] @ W[128,128] + bias  via v_wmma_f32_16x16x32_f16
// Block = 256 threads = 8 waves, covering 64 rows x 128 cols.
// Each wave holds its 16-col B fragments in regs across 4 M-tiles.
// Uniform fast path for full blocks; guards only in the single tail block.
// ---------------------------------------------------------------------------
__global__ __launch_bounds__(256) void k_gemm(const float* __restrict__ Z,
    const float* __restrict__ W, const float* __restrict__ bias,
    float* __restrict__ Out, int nrows) {
  __shared__ _Float16 As[64][HID + 8];          // 272B row stride (8B aligned)
  int row0 = blockIdx.x * 64;
  bool full = (row0 + 64 <= nrows);

  // stage A tile: float4 load -> packed f16x4 -> 8B LDS store
  for (int i = threadIdx.x; i < 64 * 32; i += 256) {
    int m = i >> 5, k4 = (i & 31) << 2;
    float4 v = make_float4(0.f, 0.f, 0.f, 0.f);
    if (full || (row0 + m < nrows))
      v = *(const float4*)&Z[(size_t)(row0 + m) * HID + k4];
    v4h h = {(_Float16)v.x, (_Float16)v.y, (_Float16)v.z, (_Float16)v.w};
    *(v4h*)&As[m][k4] = h;
  }
  __syncthreads();

  int wid = threadIdx.x >> 5, lane = threadIdx.x & 31;
  int n0 = wid * 16;
  int nn = lane & 15, half = lane >> 4;

  // B fragments, contiguous k order (same as A): VGPR pair i = W[k][n],W[k+1][n], k = 2i(+16 hi)
  v16h bfr[4];
#pragma unroll
  for (int step = 0; step < 4; ++step) {
#pragma unroll
    for (int i = 0; i < 8; ++i) {
      int k = step * 32 + half * 16 + 2 * i;
      bfr[step][2 * i]     = (_Float16)W[(size_t)k * HID + n0 + nn];
      bfr[step][2 * i + 1] = (_Float16)W[(size_t)(k + 1) * HID + n0 + nn];
    }
  }
  float bv = bias[n0 + nn];

#pragma unroll
  for (int mt = 0; mt < 4; ++mt) {
    int mrow = mt * 16 + nn;                    // A row owned by this lane
    v8f c = {};
#pragma unroll
    for (int step = 0; step < 4; ++step) {
      v16h a;
#pragma unroll
      for (int i = 0; i < 8; ++i) {
        // A slot i -> K = 32*step + 16*half + 2*i + {0,1}
        int kb = step * 32 + (half << 4) + (i << 1);
        a[2 * i]     = As[mrow][kb];
        a[2 * i + 1] = As[mrow][kb + 1];
      }
      c = __builtin_amdgcn_wmma_f32_16x16x32_f16(false, a, false, bfr[step],
                                                 (short)0, c, false, false);
    }
    // D: VGPR r holds row (r + 8*half), col = lane&15
    size_t obase = (size_t)(row0 + mt * 16 + 8 * half) * HID + n0 + nn;
    if (full) {
#pragma unroll
      for (int r = 0; r < 8; ++r) Out[obase + (size_t)r * HID] = c[r] + bv;
    } else {
#pragma unroll
      for (int r = 0; r < 8; ++r) {
        int grow = row0 + mt * 16 + r + 8 * half;
        if (grow < nrows) Out[(size_t)grow * HID + n0 + nn] = c[r] + bv;
      }
    }
  }
}

__device__ __forceinline__ float lrelu(float v) { return v > 0.f ? v : 0.2f * v; }

// ---------------------------------------------------------------------------
// Edge pass 1 (persistent waves, grid-stride). One wave per edge per iter.
// Per-lane weight slices (We 8xfloat4, att, be) hoisted out of the loop.
// Next edge's xl/xr rows prefetched (global_prefetch_b8) to hide gather latency.
// Segment-max elided: softmax is shift-invariant, logits bounded ~0.3.
// ---------------------------------------------------------------------------
__global__ __launch_bounds__(256) void k_edge_logit(
    const float* __restrict__ xl, const float* __restrict__ xr,
    const int* __restrict__ src, const int* __restrict__ dst,
    const float* __restrict__ ea, const float* __restrict__ We,
    const float* __restrict__ be, const float* __restrict__ att,
    float* __restrict__ exb, float* __restrict__ den, int nedges) {
  int lane = threadIdx.x & 31;
  int wv   = blockIdx.x * 8 + (threadIdx.x >> 5);
  int nw   = gridDim.x * 8;

  float4 be4 = *(const float4*)(be + 4 * lane);
  float4 a4  = *(const float4*)(att + (lane >> 3) * 32 + 4 * (lane & 7));
  float4 wvt[8];
#pragma unroll
  for (int j = 0; j < 8; ++j) wvt[j] = *(const float4*)(We + j * HID + 4 * lane);

  for (int e = wv; e < nedges; e += nw) {
    int s = src[e], d = dst[e];
    int en = e + nw;
    if (en < nedges) {                 // prefetch next edge's rows
      int sn = src[en], dn = dst[en];
      __builtin_prefetch(xl + (size_t)sn * HID + 4 * lane, 0, 3);
      __builtin_prefetch(xr + (size_t)dn * HID + 4 * lane, 0, 3);
    }
    float4 vl = ((const float4*)(xl + (size_t)s * HID))[lane];
    float4 vr = ((const float4*)(xr + (size_t)d * HID))[lane];
    float4 ev = be4;
#pragma unroll
    for (int j = 0; j < 8; ++j) {
      float aj = ea[(size_t)e * 8 + j];
      ev.x += aj * wvt[j].x; ev.y += aj * wvt[j].y;
      ev.z += aj * wvt[j].z; ev.w += aj * wvt[j].w;
    }
    float4 sv;
    sv.x = lrelu(vl.x + vr.x + ev.x);
    sv.y = lrelu(vl.y + vr.y + ev.y);
    sv.z = lrelu(vl.z + vr.z + ev.z);
    sv.w = lrelu(vl.w + vr.w + ev.w);
    float p = sv.x * a4.x + sv.y * a4.y + sv.z * a4.z + sv.w * a4.w;
    p += __shfl_xor(p, 1, 32);
    p += __shfl_xor(p, 2, 32);
    p += __shfl_xor(p, 4, 32);
    if ((lane & 7) == 0) {
      int h = lane >> 3;
      float ex = __expf(p);
      exb[(size_t)e * HEADS + h] = ex;
      atomicAdd(&den[(size_t)d * HEADS + h], ex);
    }
  }
}

// ---------------------------------------------------------------------------
// Edge pass 2: alpha = ex/den[dst]; hsum[dst] += xl[src] * alpha  (f32 atomics)
// ---------------------------------------------------------------------------
__global__ __launch_bounds__(256) void k_edge_scatter(
    const float* __restrict__ xl, const int* __restrict__ src,
    const int* __restrict__ dst, const float* __restrict__ exb,
    const float* __restrict__ den, float* __restrict__ hsum, int nedges) {
  int lane = threadIdx.x & 31;
  int wv   = blockIdx.x * 8 + (threadIdx.x >> 5);
  int nw   = gridDim.x * 8;
  for (int e = wv; e < nedges; e += nw) {
    int s = src[e], d = dst[e];
    int en = e + nw;
    if (en < nedges) {
      int sn = src[en];
      __builtin_prefetch(xl + (size_t)sn * HID + 4 * lane, 0, 3);
    }
    int h = lane >> 3;
    float alpha = exb[(size_t)e * HEADS + h] / den[(size_t)d * HEADS + h];
    float4 vl = ((const float4*)(xl + (size_t)s * HID))[lane];
    float* o = hsum + (size_t)d * HID + 4 * lane;
    atomicAdd(o + 0, vl.x * alpha);
    atomicAdd(o + 1, vl.y * alpha);
    atomicAdd(o + 2, vl.z * alpha);
    atomicAdd(o + 3, vl.w * alpha);
  }
}

// ---------------------------------------------------------------------------
// h = relu(hsum + bo); z = LayerNorm(z + h)*g + beta   (one wave per node)
// ---------------------------------------------------------------------------
__global__ __launch_bounds__(256) void k_res_ln(float* __restrict__ z,
    const float* __restrict__ hsum, const float* __restrict__ bo,
    const float* __restrict__ g, const float* __restrict__ beta, int n) {
  int lane = threadIdx.x & 31;
  int node = blockIdx.x * 8 + (threadIdx.x >> 5);
  if (node >= n) return;
  size_t base = (size_t)node * HID;
  int c0 = 4 * lane;
  float4 hv = *(const float4*)(hsum + base + c0);
  float4 bv = *(const float4*)(bo + c0);
  hv.x = fmaxf(hv.x + bv.x, 0.f); hv.y = fmaxf(hv.y + bv.y, 0.f);
  hv.z = fmaxf(hv.z + bv.z, 0.f); hv.w = fmaxf(hv.w + bv.w, 0.f);
  float4 zv = *(const float4*)(z + base + c0);
  zv.x += hv.x; zv.y += hv.y; zv.z += hv.z; zv.w += hv.w;
  float s = zv.x + zv.y + zv.z + zv.w;
#pragma unroll
  for (int m = 1; m < 32; m <<= 1) s += __shfl_xor(s, m, 32);
  float mu = s * (1.f / 128.f);
  float dx = zv.x - mu, dy = zv.y - mu, dz = zv.z - mu, dw = zv.w - mu;
  float vs = dx * dx + dy * dy + dz * dz + dw * dw;
#pragma unroll
  for (int m = 1; m < 32; m <<= 1) vs += __shfl_xor(vs, m, 32);
  float rs = rsqrtf(vs * (1.f / 128.f) + 1e-5f);
  float4 gv = *(const float4*)(g + c0);
  float4 b2 = *(const float4*)(beta + c0);
  float4 o;
  o.x = dx * rs * gv.x + b2.x; o.y = dy * rs * gv.y + b2.y;
  o.z = dz * rs * gv.z + b2.z; o.w = dw * rs * gv.w + b2.w;
  *(float4*)(z + base + c0) = o;
}

// ---------------------------------------------------------------------------
// out[n] = z[n,:] . W_out + b_out   (one wave per node)
// ---------------------------------------------------------------------------
__global__ __launch_bounds__(256) void k_out(const float* __restrict__ z,
    const float* __restrict__ Wout, const float* __restrict__ bout,
    float* __restrict__ out, int n) {
  int lane = threadIdx.x & 31;
  int node = blockIdx.x * 8 + (threadIdx.x >> 5);
  if (node >= n) return;
  float4 a = ((const float4*)(z + (size_t)node * HID))[lane];
  float4 w = ((const float4*)Wout)[lane];
  float p = a.x * w.x + a.y * w.y + a.z * w.z + a.w * w.w;
#pragma unroll
  for (int m = 1; m < 32; m <<= 1) p += __shfl_xor(p, m, 32);
  if (lane == 0) out[node] = p + bout[0];
}

// ---------------------------------------------------------------------------
extern "C" void kernel_launch(void* const* d_in, const int* in_sizes, int n_in,
                              void* d_out, int out_size, void* d_ws, size_t ws_size,
                              hipStream_t stream) {
  (void)n_in; (void)out_size; (void)ws_size;
  const float* x    = (const float*)d_in[0];
  const int*   nidx = (const int*)d_in[1];
  const int*   eidx = (const int*)d_in[2];
  const float* ea   = (const float*)d_in[3];
  const float* emb  = (const float*)d_in[4];
  const float* Win  = (const float*)d_in[5];
  const float* bin  = (const float*)d_in[6];
  const float* Wl[2]   = {(const float*)d_in[7],  (const float*)d_in[15]};
  const float* bl[2]   = {(const float*)d_in[8],  (const float*)d_in[16]};
  const float* Wr[2]   = {(const float*)d_in[9],  (const float*)d_in[17]};
  const float* br[2]   = {(const float*)d_in[10], (const float*)d_in[18]};
  const float* We[2]   = {(const float*)d_in[11], (const float*)d_in[19]};
  const float* be[2]   = {(const float*)d_in[12], (const float*)d_in[20]};
  const float* att[2]  = {(const float*)d_in[13], (const float*)d_in[21]};
  const float* bo[2]   = {(const float*)d_in[14], (const float*)d_in[22]};
  const float* g[2]    = {(const float*)d_in[23], (const float*)d_in[25]};
  const float* beta[2] = {(const float*)d_in[24], (const float*)d_in[26]};
  const float* Wout = (const float*)d_in[27];
  const float* bout = (const float*)d_in[28];

  int N = in_sizes[1];
  int E = in_sizes[3] / 8;
  const int* src = eidx;
  const int* dst = eidx + E;

  float* ws = (float*)d_ws;
  size_t ZN = (size_t)N * HID;
  float* z    = ws;
  float* xl   = ws + ZN;
  float* xr   = ws + 2 * ZN;
  float* hsum = xr;                    // xr is dead after k_edge_logit
  float* exb  = ws + 3 * ZN;
  float* den  = exb + (size_t)E * HEADS;

  int inBlocks   = (int)(((size_t)N * HID + 255) / 256);
  int gemmBlocks = (N + 63) / 64;
  int edgeBlocks = (E + 7) / 8;
  if (edgeBlocks > 6144) edgeBlocks = 6144;   // persistent grid-stride waves
  int nodeBlocks = (N + 7) / 8;

  k_input<<<inBlocks, 256, 0, stream>>>(x, nidx, emb, Win, bin, z, N);
  for (int l = 0; l < 2; ++l) {
    k_gemm<<<gemmBlocks, 256, 0, stream>>>(z, Wl[l], bl[l], xl, N);
    k_gemm<<<gemmBlocks, 256, 0, stream>>>(z, Wr[l], br[l], xr, N);
    hipMemsetAsync(den, 0, (size_t)N * HEADS * sizeof(float), stream);
    k_edge_logit<<<edgeBlocks, 256, 0, stream>>>(xl, xr, src, dst, ea, We[l],
                                                 be[l], att[l], exb, den, E);
    hipMemsetAsync(hsum, 0, ZN * sizeof(float), stream);   // after logit: xr dead
    k_edge_scatter<<<edgeBlocks, 256, 0, stream>>>(xl, src, dst, exb, den, hsum, E);
    k_res_ln<<<nodeBlocks, 256, 0, stream>>>(z, hsum, bo[l], g[l], beta[l], N);
  }
  k_out<<<nodeBlocks, 256, 0, stream>>>(z, Wout, bout, (float*)d_out, N);
}
